// MPNN_61890478735948
// MI455X (gfx1250) — matrix operand order repaired and checked by reading, not compiled, not run on hardware
//
#include <hip/hip_runtime.h>
#include <math.h>

// ---------------------------------------------------------------------------
// MPNN forward + backward, fully fused per-atom, CDNA5 (gfx1250) WMMA bf16.
// Sizes fixed by the reference: B=8, A=256, n=64, NF=128, RES=20, CUT=5, P=9.
// Input order assumes jax-style alphabetical dict flattening:
//  0:R 1:Z 2:N 3:NM 4:atom.W1 5:atom.W2 6:atom.W3 7:atom.b1 8:atom.b2 9:atom.b3
//  10:emb 11:Wa1 12:Wa2 13:Wr 14:ba1 15:ba2 16:br
//  17:pair.W1 18:pair.W2 19:pair.W3 20:pair.b1 21:pair.b2 22:pair.b3
// Output: atom_pred[2048] | pair_pred[131072] | GAi[6144] | GPij[131072]
// ---------------------------------------------------------------------------

#define NNB    64
#define NF_    128
#define NRES   20
#define NA_    256
#define NB_    8
#define NROW   (NB_*NA_)          // 2048
#define CUTR   5.0f
#define PI_F   3.14159265358979323846f

typedef __attribute__((ext_vector_type(16))) __bf16 v16bf;
typedef __attribute__((ext_vector_type(8)))  float  v8f;
typedef unsigned short u16;

static __device__ inline u16  f2bf(float x) { return __builtin_bit_cast(u16, (__bf16)x); }
static __device__ inline float bf2f(u16 u)  { return (float)__builtin_bit_cast(__bf16, u); }
static __device__ inline float sigf(float x){ return 1.0f / (1.0f + expf(-x)); }
static __device__ inline float silupf(float z){ float s = sigf(z); return s*(1.0f + z*(1.0f - s)); }

static __device__ inline v8f vzero() {
  v8f z;
#pragma unroll
  for (int i = 0; i < 8; ++i) z[i] = 0.0f;
  return z;
}

// A fragment (16x32 bf16, M x K) from LDS, row-major [row][k], leading dim ld.
// Layout (ISA 7.12.2): lanes 0-15 row M=lane: h0..7 -> K=0..7, h8..15 -> K=16..23
//                      lanes 16-31 row M=lane-16: h0..7 -> K=8..15, h8..15 -> K=24..31
static __device__ inline v16bf lda(const u16* Ab, int ld, int mt, int kb, int lane) {
  v16bf a;
  int row = mt * 16 + (lane & 15);
  int k0  = kb + ((lane & 16) ? 8 : 0);
  const u16* p = Ab + row * ld + k0;
#pragma unroll
  for (int h = 0; h < 8; ++h) a[h] = __builtin_bit_cast(__bf16, p[h]);
#pragma unroll
  for (int h = 0; h < 8; ++h) a[h + 8] = __builtin_bit_cast(__bf16, p[16 + h]);
  return a;
}

// B fragment (32x16 bf16, K x N) from a bf16 array stored [n][k], leading dim ld.
// Layout: lanes 0-15: N=lane, h0..7 -> K=0..7, h8..15 -> K=16..23 ; lanes 16-31: N=lane-16, h0..7 -> K=8..15, h8..15 -> K=24..31.
static __device__ inline v16bf ldb(const u16* Bt, int ld, int nt, int kb, int lane) {
  v16bf b;
  int n  = nt * 16 + (lane & 15);
  int k0 = kb + ((lane & 16) ? 8 : 0);
  const u16* p = Bt + n * ld + k0;
#pragma unroll
  for (int h = 0; h < 8; ++h) b[h] = __builtin_bit_cast(__bf16, p[h]);
#pragma unroll
  for (int h = 0; h < 8; ++h) b[h + 8] = __builtin_bit_cast(__bf16, p[16 + h]);
  return b;
}

static __device__ inline v8f wmma_bf(v16bf a, v16bf b, v8f c) {
  return __builtin_amdgcn_wmma_f32_16x16x32_bf16(false, a, false, b, (short)0, c, false, false);
}

// ---------------------------------------------------------------------------
// Weight prep: bf16 copies. *t arrays are transposed to [n][k] (B-fragment
// wants contiguous K per lane); *b arrays keep original [k_in][n_out] layout
// which is exactly the [n'][k'] layout needed by the backward GEMMs.
// ---------------------------------------------------------------------------
__global__ void k_prep(const float* Wa1, const float* Wa2, const float* Wr,
                       const float* pW1, const float* pW2,
                       u16* Wa1t, u16* Wa2t, u16* W1pt, u16* W1pb,
                       u16* W2pt, u16* W2pb, u16* Wrt, u16* Wrb) {
  int gid = blockIdx.x * blockDim.x + threadIdx.x;
  int stride = gridDim.x * blockDim.x;
  for (int i = gid; i < 16384; i += stride) {
    int n = i >> 7, k = i & 127;
    Wa1t[i] = f2bf(Wa1[k * 128 + n]);
    Wa2t[i] = f2bf(Wa2[k * 128 + n]);
    W1pt[i] = f2bf(pW1[k * 128 + n]);
    W1pb[i] = f2bf(pW1[i]);
  }
  for (int i = gid; i < 8192; i += stride) {
    int c = i >> 7, h = i & 127;
    W2pt[i] = f2bf(pW2[h * 64 + c]);   // [c][h]
    W2pb[i] = f2bf(pW2[i]);            // [h][c]
  }
  for (int i = gid; i < 4096; i += stride) {
    int n = i >> 5, k = i & 31;        // Wrt: [f][r], K padded 20->32
    Wrt[i] = (k < NRES) ? f2bf(Wr[k * 128 + n]) : (u16)0;
    int r = i >> 7, f = i & 127;       // Wrb: [r][f], rows padded 20->32
    Wrb[i] = (r < NRES) ? f2bf(Wr[r * 128 + f]) : (u16)0;
  }
}

__global__ void k_zero(float* p, int n) {
  int i = blockIdx.x * blockDim.x + threadIdx.x;
  if (i < n) p[i] = 0.0f;
}

// ---------------------------------------------------------------------------
// am = silu(a0 @ Wa1 + ba1) @ Wa2 + ba2 for all 2048 atom rows.
// ---------------------------------------------------------------------------
__global__ __launch_bounds__(256) void k_am(const int* Zt, const float* emb,
    const u16* Wa1t, const u16* Wa2t, const float* ba1, const float* ba2,
    float* am) {
  __shared__ u16 sAb[64 * 128];
  const int tid = threadIdx.x, wave = tid >> 5, lane = tid & 31;
  const int row0 = blockIdx.x * 64;

  { int f = tid & 127, half = tid >> 7;
    for (int rr = half * 32; rr < half * 32 + 32; ++rr) {
      int zt = Zt[row0 + rr];
      sAb[rr * 128 + f] = f2bf(emb[zt * 128 + f]);
    } }
  __syncthreads();

  float hh[4][8];
  { int ti = 0;
    for (int t = wave; t < 32; t += 8, ++ti) {
      int mt = t & 3, nt = t >> 2;
      v8f acc = vzero();
#pragma unroll
      for (int kb = 0; kb < 128; kb += 32)
        acc = wmma_bf(lda(sAb, 128, mt, kb, lane), ldb(Wa1t, 128, nt, kb, lane), acc);
      int col = nt * 16 + (lane & 15);
      float bias = ba1[col];
#pragma unroll
      for (int r = 0; r < 8; ++r) { float z = acc[r] + bias; hh[ti][r] = z * sigf(z); }
    } }
  __syncthreads();
  { int ti = 0;
    for (int t = wave; t < 32; t += 8, ++ti) {
      int mt = t & 3, nt = t >> 2;
      int col = nt * 16 + (lane & 15);
      int rb = mt * 16 + ((lane & 16) ? 8 : 0);
#pragma unroll
      for (int r = 0; r < 8; ++r) sAb[(rb + r) * 128 + col] = f2bf(hh[ti][r]);
    } }
  __syncthreads();
  { for (int t = wave; t < 32; t += 8) {
      int mt = t & 3, nt = t >> 2;
      v8f acc = vzero();
#pragma unroll
      for (int kb = 0; kb < 128; kb += 32)
        acc = wmma_bf(lda(sAb, 128, mt, kb, lane), ldb(Wa2t, 128, nt, kb, lane), acc);
      int col = nt * 16 + (lane & 15);
      int rb = mt * 16 + ((lane & 16) ? 8 : 0);
      float bias = ba2[col];
#pragma unroll
      for (int r = 0; r < 8; ++r) am[(row0 + rb + r) * 128 + col] = acc[r] + bias;
    } }
}

// ---------------------------------------------------------------------------
// Fused per-atom forward + backward. One block = one atom (64 edges x 128 f).
// ---------------------------------------------------------------------------
__global__ __launch_bounds__(256) void k_main(
    const float* Rp, const int* Zt, const int* Nbr, const float* NMp,
    const float* emb, const float* am,
    const u16* Wrt, const u16* Wrb, const float* br,
    const u16* W1pt, const u16* W1pb, const float* pb1,
    const u16* W2pt, const u16* W2pb, const float* pb2,
    const float* pW3, const float* pb3,
    const float* aW1, const float* aW2, const float* aW3,
    const float* ab1, const float* ab2, const float* ab3,
    float* out_atom, float* out_pair, float* out_gai, float* out_gpij) {

  __shared__ u16 sAb[64 * 128];     // rotating A operand (rbf,p,h1,g_z2,g_z1,gm)
  __shared__ u16 sPre[64 * 128];    // pre = rbf@Wr + br   (bf16)
  __shared__ u16 sSig1[64 * 128];   // silu'(z1)           (bf16)
  __shared__ float sD[64], sDraw[64], sNM[64], sCF[64], sTap[64];
  __shared__ float sQ[64], sGq[64], sGcfP[64], sGcfA[64], sGDp[64], sGDa[64];
  __shared__ float sV[64 * 3];
  __shared__ int   sNbr[64], sZj[64];
  __shared__ float sA0i[128], sAmI[128], sA1[128], sGA1[128];
  __shared__ float sA1part[256];
  __shared__ float sZ1a[128], sH1a[128], sZ2a[64], sGH2a[64], sGH1a[128];
  __shared__ float sRed[64];
  __shared__ float sGAiI[3];

  const int tid = threadIdx.x, wave = tid >> 5, lane = tid & 31;
  const int row = blockIdx.x;          // global atom index b*256+a
  const int bb = row >> 8;             // batch

  // ---- phase 0a: per-edge geometry, cutoff, tap; gathers; accumulator init
  if (tid < NNB) {
    int e = tid;
    int j = Nbr[row * NNB + e];
    sNbr[e] = j;
    sZj[e] = Zt[bb * NA_ + j];
    float nm = NMp[row * NNB + e];
    sNM[e] = nm;
    float d2 = 0.0f;
#pragma unroll
    for (int c = 0; c < 3; ++c) {
      float v = Rp[(bb * NA_ + j) * 3 + c] - Rp[row * 3 + c];
      sV[e * 3 + c] = v;
      d2 += v * v;
    }
    float draw = sqrtf(fmaxf(d2, 1e-12f));
    float D = draw * nm;
    sDraw[e] = draw; sD[e] = D;
    float x = D * (1.0f / CUTR);
    float cf = 0.0f;
    if (x < 1.0f) {
      float x2 = x * x, x4 = x2 * x2, x8 = x4 * x4;
      float x9 = x8 * x, x10 = x9 * x, x11 = x10 * x;
      cf = 1.0f - 55.0f * x9 + 99.0f * x10 - 45.0f * x11;
    }
    sCF[e] = cf;
    sTap[e] = 0.5f + 0.5f * tanhf(10.0f * (5.0f - D));
    sQ[e] = 0.0f; sGcfP[e] = 0.0f; sGcfA[e] = 0.0f; sGDp[e] = 0.0f; sGDa[e] = 0.0f;
  }
  if (tid >= 64 && tid < 192) {
    int f = tid - 64;
    sA0i[f] = emb[Zt[row] * NF_ + f];
    sAmI[f] = am[row * NF_ + f];
  }
  sA1part[tid] = 0.0f;
  if (tid < 3) sGAiI[tid] = 0.0f;
  __syncthreads();

  // ---- phase 0b: bessel RBF into sAb as 64x32 bf16 (cols 20..31 zero)
  for (int idx = tid; idx < NNB * 32; idx += 256) {
    int e = idx >> 5, r = idx & 31;
    float D = sD[e];
    u16 v = 0;
    if (r < NRES && D > 0.0f) {
      float fr = (float)(r + 1) * (PI_F / CUTR);
      v = f2bf(sinf(fr * D) / D);
    }
    sAb[e * 32 + r] = v;
  }
  __syncthreads();

  // ---- G1: pre = rbf @ Wr + br   (M=64,K=32,N=128) -> sPre (bf16)
  for (int t = wave; t < 32; t += 8) {
    int mt = t & 3, nt = t >> 2;
    v8f acc = vzero();
    acc = wmma_bf(lda(sAb, 32, mt, 0, lane), ldb(Wrt, 32, nt, 0, lane), acc);
    int col = nt * 16 + (lane & 15);
    int rb = mt * 16 + ((lane & 16) ? 8 : 0);
    float bias = br[col];
#pragma unroll
    for (int r = 0; r < 8; ++r) sPre[(rb + r) * NF_ + col] = f2bf(acc[r] + bias);
  }
  __syncthreads();

  // ---- E1: p = a0_i*a0_j + msij (-> sAb bf16), partial a1 = sum_e msij
  { int f = tid & 127, half = tid >> 7;
    float a0i = sA0i[f], ami = sAmI[f];
    float part = 0.0f;
    for (int e = half * 32; e < half * 32 + 32; ++e) {
      float pre = bf2f(sPre[e * NF_ + f]);
      int j = sNbr[e];
      float w = ami * am[j * NF_ + f] * sNM[e];
      float ms = pre * sCF[e] * w;
      float p = a0i * emb[sZj[e] * NF_ + f] + ms;
      sAb[e * NF_ + f] = f2bf(p);
      part += ms;
    }
    sA1part[half * NF_ + f] = part;
  }
  __syncthreads();
  if (tid < NF_) sA1[tid] = sA0i[tid] + sA1part[tid] + sA1part[NF_ + tid];
  __syncthreads();

  // ---- atom MLP forward + backward (1x128 row; plain VALU GEMVs)
  if (tid < NF_) {
    int o = tid;
    float acc = ab1[o];
    for (int k = 0; k < NF_; ++k) acc += sA1[k] * aW1[k * 128 + o];
    sZ1a[o] = acc;
    sH1a[o] = acc * sigf(acc);
  }
  __syncthreads();
  if (tid < 64) {
    int c = tid;
    float acc = ab2[c];
    for (int k = 0; k < NF_; ++k) acc += sH1a[k] * aW2[k * 64 + c];
    sZ2a[c] = acc;
    float h2 = acc * sigf(acc);
    sRed[c] = h2 * aW3[c];
  }
  __syncthreads();
  if (tid == 0) {
    float s = ab3[0];
    for (int c = 0; c < 64; ++c) s += sRed[c];
    out_atom[row] = s;
  }
  if (tid < 64) sGH2a[tid] = aW3[tid] * silupf(sZ2a[tid]);
  __syncthreads();
  if (tid < NF_) {
    int k = tid;
    float acc = 0.0f;
    for (int c = 0; c < 64; ++c) acc += sGH2a[c] * aW2[k * 64 + c];
    sGH1a[k] = acc * silupf(sZ1a[k]);
  }
  __syncthreads();
  if (tid < NF_) {
    int f = tid;
    float acc = 0.0f;
    for (int k = 0; k < NF_; ++k) acc += sGH1a[k] * aW1[f * 128 + k];
    sGA1[f] = acc;
  }
  __syncthreads();

  // ---- G2: z1 = p @ W1 + b1 ; h1=silu(z1) -> sAb ; silu'(z1) -> sSig1
  float hh[4][8], hsp[4][8];
  { int ti = 0;
    for (int t = wave; t < 32; t += 8, ++ti) {
      int mt = t & 3, nt = t >> 2;
      v8f acc = vzero();
#pragma unroll
      for (int kb = 0; kb < 128; kb += 32)
        acc = wmma_bf(lda(sAb, 128, mt, kb, lane), ldb(W1pt, 128, nt, kb, lane), acc);
      int col = nt * 16 + (lane & 15);
      float bias = pb1[col];
#pragma unroll
      for (int r = 0; r < 8; ++r) {
        float z = acc[r] + bias;
        float s = sigf(z);
        hh[ti][r] = z * s;
        hsp[ti][r] = s * (1.0f + z * (1.0f - s));
      }
    } }
  __syncthreads();
  { int ti = 0;
    for (int t = wave; t < 32; t += 8, ++ti) {
      int mt = t & 3, nt = t >> 2;
      int col = nt * 16 + (lane & 15);
      int rb = mt * 16 + ((lane & 16) ? 8 : 0);
#pragma unroll
      for (int r = 0; r < 8; ++r) {
        sAb[(rb + r) * NF_ + col] = f2bf(hh[ti][r]);
        sSig1[(rb + r) * NF_ + col] = f2bf(hsp[ti][r]);
      }
    } }
  __syncthreads();

  // ---- G3: z2 = h1 @ W2 + b2 (N=64); q partials ; keep silu'(z2) in regs
  float hsp2[2][8];
  { int ti = 0;
    for (int t = wave; t < 16; t += 8, ++ti) {
      int mt = t & 3, nt = t >> 2;
      v8f acc = vzero();
#pragma unroll
      for (int kb = 0; kb < 128; kb += 32)
        acc = wmma_bf(lda(sAb, 128, mt, kb, lane), ldb(W2pt, 128, nt, kb, lane), acc);
      int col = nt * 16 + (lane & 15);
      int rb = mt * 16 + ((lane & 16) ? 8 : 0);
      float bias = pb2[col], w3 = pW3[col];
#pragma unroll
      for (int r = 0; r < 8; ++r) {
        float z = acc[r] + bias;
        float s = sigf(z);
        hsp2[ti][r] = s * (1.0f + z * (1.0f - s));
        atomicAdd(&sQ[rb + r], (z * s) * w3);
      }
    } }
  __syncthreads();
  if (tid < NNB) {
    int e = tid;
    float q = sQ[e] + pb3[0];
    float mask = (sD[e] != 0.0f) ? 1.0f : 0.0f;
    out_pair[row * NNB + e] = q * q * mask * sTap[e];
    sGq[e] = 2.0f * q * mask * sTap[e];
  }
  __syncthreads();
  // g_z2 = g_q * W3 * silu'(z2) -> sAb (64x64 region, ld 128)
  { int ti = 0;
    for (int t = wave; t < 16; t += 8, ++ti) {
      int mt = t & 3, nt = t >> 2;
      int col = nt * 16 + (lane & 15);
      int rb = mt * 16 + ((lane & 16) ? 8 : 0);
      float w3 = pW3[col];
#pragma unroll
      for (int r = 0; r < 8; ++r)
        sAb[(rb + r) * NF_ + col] = f2bf(sGq[rb + r] * w3 * hsp2[ti][r]);
    } }
  __syncthreads();

  // ---- G4: g_z1 = (g_z2 @ W2^T) * silu'(z1)   (M=64,K=64,N=128)
  { int ti = 0;
    for (int t = wave; t < 32; t += 8, ++ti) {
      int mt = t & 3, nt = t >> 2;
      v8f acc = vzero();
#pragma unroll
      for (int kb = 0; kb < 64; kb += 32)
        acc = wmma_bf(lda(sAb, 128, mt, kb, lane), ldb(W2pb, 64, nt, kb, lane), acc);
      int col = nt * 16 + (lane & 15);
      int rb = mt * 16 + ((lane & 16) ? 8 : 0);
#pragma unroll
      for (int r = 0; r < 8; ++r)
        hh[ti][r] = acc[r] * bf2f(sSig1[(rb + r) * NF_ + col]);
    } }
  __syncthreads();
  { int ti = 0;
    for (int t = wave; t < 32; t += 8, ++ti) {
      int mt = t & 3, nt = t >> 2;
      int col = nt * 16 + (lane & 15);
      int rb = mt * 16 + ((lane & 16) ? 8 : 0);
#pragma unroll
      for (int r = 0; r < 8; ++r) sAb[(rb + r) * NF_ + col] = f2bf(hh[ti][r]);
    } }
  __syncthreads();

  // ---- G5: g_p = g_z1 @ W1^T ; gm_pair = g_p*w*cf -> sAb ; g_cf_pair accum
  { int ti = 0;
    for (int t = wave; t < 32; t += 8, ++ti) {
      int mt = t & 3, nt = t >> 2;
      v8f acc = vzero();
#pragma unroll
      for (int kb = 0; kb < 128; kb += 32)
        acc = wmma_bf(lda(sAb, 128, mt, kb, lane), ldb(W1pb, 128, nt, kb, lane), acc);
      int col = nt * 16 + (lane & 15);
      int rb = mt * 16 + ((lane & 16) ? 8 : 0);
      float ami = sAmI[col];
#pragma unroll
      for (int r = 0; r < 8; ++r) {
        int e = rb + r;
        float w = ami * am[sNbr[e] * NF_ + col] * sNM[e];
        float gp = acc[r];
        atomicAdd(&sGcfP[e], gp * w * bf2f(sPre[e * NF_ + col]));
        hh[ti][r] = gp * w * sCF[e];
      }
    } }
  __syncthreads();
  { int ti = 0;
    for (int t = wave; t < 32; t += 8, ++ti) {
      int mt = t & 3, nt = t >> 2;
      int col = nt * 16 + (lane & 15);
      int rb = mt * 16 + ((lane & 16) ? 8 : 0);
#pragma unroll
      for (int r = 0; r < 8; ++r) sAb[(rb + r) * NF_ + col] = f2bf(hh[ti][r]);
    } }
  __syncthreads();

  // ---- G6: g_rbf_pair = gm_pair @ Wr^T ; dD accumulation via dBessel/dD
  { int t = wave;                  // 8 tiles: 4 mt x 2 nt
    int mt = t & 3, nt = t >> 2;
    v8f acc = vzero();
#pragma unroll
    for (int kb = 0; kb < 128; kb += 32)
      acc = wmma_bf(lda(sAb, 128, mt, kb, lane), ldb(Wrb, 128, nt, kb, lane), acc);
    int rcol = nt * 16 + (lane & 15);
    int rb = mt * 16 + ((lane & 16) ? 8 : 0);
    if (rcol < NRES) {
      float fr = (float)(rcol + 1) * (PI_F / CUTR);
#pragma unroll
      for (int r = 0; r < 8; ++r) {
        int e = rb + r;
        float d = sD[e];
        if (d > 0.0f) {
          float drb = (fr * cosf(fr * d) * d - sinf(fr * d)) / (d * d);
          atomicAdd(&sGDp[e], acc[r] * drb);
        }
      }
    } }
  __syncthreads();

  // ---- E2: gm_atom = g_a1*w*cf -> sAb ; g_cf_atom accumulation
  { int e = tid >> 2, qq = tid & 3;
    int j = sNbr[e];
    float nm = sNM[e], cf = sCF[e];
    float acc = 0.0f;
    for (int f = qq * 32; f < qq * 32 + 32; ++f) {
      float w = sAmI[f] * am[j * NF_ + f] * nm;
      float g = sGA1[f] * w;
      sAb[e * NF_ + f] = f2bf(g * cf);
      acc += g * bf2f(sPre[e * NF_ + f]);
    }
    atomicAdd(&sGcfA[e], acc);
  }
  __syncthreads();

  // ---- G7: g_rbf_atom = gm_atom @ Wr^T ; dD accumulation
  { int t = wave;
    int mt = t & 3, nt = t >> 2;
    v8f acc = vzero();
#pragma unroll
    for (int kb = 0; kb < 128; kb += 32)
      acc = wmma_bf(lda(sAb, 128, mt, kb, lane), ldb(Wrb, 128, nt, kb, lane), acc);
    int rcol = nt * 16 + (lane & 15);
    int rb = mt * 16 + ((lane & 16) ? 8 : 0);
    if (rcol < NRES) {
      float fr = (float)(rcol + 1) * (PI_F / CUTR);
#pragma unroll
      for (int r = 0; r < 8; ++r) {
        int e = rb + r;
        float d = sD[e];
        if (d > 0.0f) {
          float drb = (fr * cosf(fr * d) * d - sinf(fr * d)) / (d * d);
          atomicAdd(&sGDa[e], acc[r] * drb);
        }
      }
    } }
  __syncthreads();

  // ---- E3: finalize GPij ; scatter GAi
  if (tid < NNB) {
    int e = tid;
    float d = sD[e];
    float x = d * (1.0f / CUTR);
    float dcf = 0.0f;
    if (x > 0.0f && x < 1.0f) {
      float x2 = x * x, x4 = x2 * x2, x8 = x4 * x4;
      dcf = (-495.0f * x8 + 990.0f * x8 * x - 495.0f * x8 * x2) * (1.0f / CUTR);
    }
    float q = sQ[e] + pb3[0];
    float mask = (d != 0.0f) ? 1.0f : 0.0f;
    float th = 2.0f * sTap[e] - 1.0f;
    float dtap = -5.0f * (1.0f - th * th);
    out_gpij[row * NNB + e] = sGDp[e] + sGcfP[e] * dcf + q * q * mask * dtap;

    float gDa = sGDa[e] + sGcfA[e] * dcf;
    float draw = sDraw[e];
    if (draw * draw > 1e-12f) {
      float fac = gDa * sNM[e] / draw;
#pragma unroll
      for (int c = 0; c < 3; ++c) {
        float gj = fac * sV[e * 3 + c];
        atomicAdd(&out_gai[(bb * NA_ + sNbr[e]) * 3 + c], gj);
        atomicAdd(&sGAiI[c], -gj);
      }
    }
  }
  __syncthreads();
  if (tid < 3) atomicAdd(&out_gai[row * 3 + tid], sGAiI[tid]);
}

// ---------------------------------------------------------------------------
extern "C" void kernel_launch(void* const* d_in, const int* in_sizes, int n_in,
                              void* d_out, int out_size, void* d_ws, size_t ws_size,
                              hipStream_t stream) {
  (void)in_sizes; (void)n_in; (void)out_size; (void)ws_size;
  const float* R   = (const float*)d_in[0];
  const int*   Zt  = (const int*)d_in[1];
  const int*   Nb  = (const int*)d_in[2];
  const float* NM  = (const float*)d_in[3];
  const float* aW1 = (const float*)d_in[4];
  const float* aW2 = (const float*)d_in[5];
  const float* aW3 = (const float*)d_in[6];
  const float* ab1 = (const float*)d_in[7];
  const float* ab2 = (const float*)d_in[8];
  const float* ab3 = (const float*)d_in[9];
  const float* emb = (const float*)d_in[10];
  const float* Wa1 = (const float*)d_in[11];
  const float* Wa2 = (const float*)d_in[12];
  const float* Wr  = (const float*)d_in[13];
  const float* ba1 = (const float*)d_in[14];
  const float* ba2 = (const float*)d_in[15];
  const float* br  = (const float*)d_in[16];
  const float* pW1 = (const float*)d_in[17];
  const float* pW2 = (const float*)d_in[18];
  const float* pW3 = (const float*)d_in[19];
  const float* pb1 = (const float*)d_in[20];
  const float* pb2 = (const float*)d_in[21];
  const float* pb3 = (const float*)d_in[22];

  char* ws = (char*)d_ws;
  u16* Wa1t = (u16*)ws;                 // 16384
  u16* Wa2t = Wa1t + 16384;
  u16* W1pt = Wa2t + 16384;
  u16* W1pb = W1pt + 16384;
  u16* W2pt = W1pb + 16384;
  u16* W2pb = W2pt + 8192;
  u16* Wrt  = W2pb + 8192;
  u16* Wrb  = Wrt + 4096;               // bf16 total 90112 elems = 180224 B
  float* am = (float*)(ws + 184320);    // 2048*128 f32 = 1 MB

  float* out = (float*)d_out;
  float* out_atom = out;                // [2048]
  float* out_pair = out + 2048;         // [131072]
  float* out_gai  = out + 133120;       // [6144]
  float* out_gpij = out + 139264;       // [131072]

  k_prep<<<64, 256, 0, stream>>>(Wa1, Wa2, Wr, pW1, pW2,
                                 Wa1t, Wa2t, W1pt, W1pb, W2pt, W2pb, Wrt, Wrb);
  k_zero<<<(6144 + 255) / 256, 256, 0, stream>>>(out_gai, 6144);
  k_am<<<NROW / 64, 256, 0, stream>>>(Zt, emb, Wa1t, Wa2t, ba1, ba2, am);
  k_main<<<NROW, 256, 0, stream>>>(R, Zt, Nb, NM, emb, am, Wrt, Wrb, br,
                                   W1pt, W1pb, pb1, W2pt, W2pb, pb2, pW3, pb3,
                                   aW1, aW2, aW3, ab1, ab2, ab3,
                                   out_atom, out_pair, out_gai, out_gpij);
}
